// INT4Linear_704374636981
// MI455X (gfx1250) — matrix operand order repaired and checked by reading, not compiled, not run on hardware
//
#include <hip/hip_runtime.h>

typedef _Float16 v16h __attribute__((ext_vector_type(16)));
typedef _Float16 v8h  __attribute__((ext_vector_type(8)));
typedef __fp16   pkh2 __attribute__((ext_vector_type(2)));   // return type of cvt_pkrtz
typedef float    v8f  __attribute__((ext_vector_type(8)));

// Problem constants (from reference setup_inputs)
constexpr int M_B   = 64;      // batch rows (M)
constexpr int K_IN  = 8192;    // in_features (K)
constexpr int N_OUT = 28672;   // out_features (N)
constexpr int GS    = 128;     // quant group size
constexpr int NGRP  = K_IN / GS;   // 64 groups
constexpr int WAVES = 4;           // 128 threads / block
constexpr int N_WG  = WAVES * 16;  // 64 output cols per block

static __device__ __forceinline__ unsigned int pack_f16x2(float a, float b) {
    pkh2 p = __builtin_amdgcn_cvt_pkrtz(a, b);
    return __builtin_bit_cast(unsigned int, p);
}

__global__ __launch_bounds__(WAVES * 32)
void int4linear_wmma_kernel(const float* __restrict__ x,
                            const int*   __restrict__ qw,      // [K/2, N] int32, low nibble=even k
                            const float* __restrict__ scales,  // [NGRP, N]
                            const int*   __restrict__ qzeros,  // [NGRP, N]
                            const float* __restrict__ bias,    // [N]
                            float*       __restrict__ out)     // [M, N]
{
    __shared__ alignas(16) _Float16 xlds[M_B * GS];   // 16 KB: x tile for one K-group, f16

    const int tid  = threadIdx.x;
    const int wave = tid >> 5;
    const int lane = tid & 31;
    const int ln   = lane & 15;       // N within tile / M within tile
    const int lh   = lane >> 4;       // half-wave select
    const int n    = blockIdx.x * N_WG + wave * 16 + ln;   // this lane's output column

    v8f acc[4] = {};   // 4 M-tiles x (16x16 f32 accumulator = 8 VGPRs)

    for (int g = 0; g < NGRP; ++g) {
        // ---- Stage x[0:64][g*128 : g*128+128] into LDS as f16 (cooperative) ----
        __syncthreads();   // previous iteration's A-fragment reads complete
        {
            const float4* xv = reinterpret_cast<const float4*>(x);
            #pragma unroll 4
            for (int i = 0; i < 16; ++i) {
                int f  = tid + i * (WAVES * 32);   // float4 index within 64x128 tile
                int m  = f >> 5;                   // 32 float4 per row
                int j4 = f & 31;
                float4 v4 = xv[(size_t)m * (K_IN / 4) + (size_t)g * (GS / 4) + j4];
                unsigned long long pk =
                    ((unsigned long long)pack_f16x2(v4.z, v4.w) << 32) |
                     (unsigned long long)pack_f16x2(v4.x, v4.y);
                *reinterpret_cast<unsigned long long*>(&xlds[m * GS + j4 * 4]) = pk;
            }
        }
        __syncthreads();

        // Per-group dequant params for this lane's column (amortized over 4 chunks)
        const float s = scales[g * N_OUT + n];
        const float z = (float)qzeros[g * N_OUT + n];

        // ---- 4 chunks of K=32 per group ----
        #pragma unroll
        for (int c = 0; c < 4; ++c) {
            // k_half row base for this lane's B fragment:
            // B k order used here: lane ln holds N=ln; lanes 0-15: K=0..15, lanes 16-31: K=16..31
            // VGPR v holds (K=2v, K=2v+1) -> exactly one packed int32 nibble pair per row.
            const int khbase = (g * GS + c * 32) / 2 + lh * 8;

            // --- Preload ALL A fragments for this chunk into distinct registers ---
            // (8 ds_load_b128 issued together; LDS latency overlaps the weight
            //  loads + dequant VALU below; then 4 WMMAs can issue back-to-back.)
            // A k order used here (matches B): lanes 0-15: K=0..15;
            // lanes 16-31: K=16..31.
            v16h afrag[4];
            #pragma unroll
            for (int mt = 0; mt < 4; ++mt) {
                const _Float16* ap = &xlds[(mt * 16 + ln) * GS + c * 32 + lh * 16];
                v8h alo = *reinterpret_cast<const v8h*>(ap);        // K  0..7 (+16*lh)
                v8h ahi = *reinterpret_cast<const v8h*>(ap + 8);    // K  8..15 (+16*lh)
                afrag[mt] = __builtin_shufflevector(alo, ahi,
                                                    0, 1, 2, 3, 4, 5, 6, 7,
                                                    8, 9, 10, 11, 12, 13, 14, 15);
            }

            // --- Stream + dequantize this lane's B fragment (weights read once) ---
            union { v16h v; unsigned int u[8]; } bf;
            #pragma unroll
            for (int r = 0; r < 8; ++r) {
                unsigned int q = (unsigned int)qw[(size_t)(khbase + r) * N_OUT + n];
                float w0 = ((float)(q & 0xFu)        - z) * s;  // even k
                float w1 = ((float)((q >> 4) & 0xFu) - z) * s;  // odd  k
                bf.u[r] = pack_f16x2(w0, w1);
            }

            // Prefetch next chunk's first weight row (global_prefetch_b8)
            if (khbase + 16 + 8 <= K_IN / 2)
                __builtin_prefetch(&qw[(size_t)(khbase + 16) * N_OUT + n], 0, 0);

            // --- 4 back-to-back WMMAs (shared B operand, independent D) ---
            #pragma unroll
            for (int mt = 0; mt < 4; ++mt) {
                acc[mt] = __builtin_amdgcn_wmma_f32_16x16x32_f16(
                    /*neg_a=*/false, afrag[mt], /*neg_b=*/false, bf.v,
                    /*c_mod=*/(short)0, acc[mt],
                    /*reuse_a=*/false, /*reuse_b=*/false);
            }
        }
    }

    // ---- Epilogue: C/D layout VGPR r -> M = r + 8*lh; add bias, store f32 ----
    const float bv = bias[n];
    #pragma unroll
    for (int mt = 0; mt < 4; ++mt) {
        #pragma unroll
        for (int r = 0; r < 8; ++r) {
            int m = mt * 16 + lh * 8 + r;
            out[(size_t)m * N_OUT + n] = acc[mt][r] + bv;
        }
    }
}

extern "C" void kernel_launch(void* const* d_in, const int* in_sizes, int n_in,
                              void* d_out, int out_size, void* d_ws, size_t ws_size,
                              hipStream_t stream) {
    (void)in_sizes; (void)n_in; (void)d_ws; (void)ws_size; (void)out_size;
    const float* x      = (const float*)d_in[0];
    const int*   qw     = (const int*)  d_in[1];
    const float* scales = (const float*)d_in[2];
    const int*   qzeros = (const int*)  d_in[3];
    const float* bias   = (const float*)d_in[4];
    // d_in[5] = group_size (128), hardcoded
    float* out = (float*)d_out;

    dim3 grid(N_OUT / N_WG);      // 448 workgroups
    dim3 block(WAVES * 32);       // 128 threads = 4 wave32
    int4linear_wmma_kernel<<<grid, block, 0, stream>>>(x, qw, scales, qzeros, bias, out);
}
